// QuantLinearW4A4_28020366639263
// MI455X (gfx1250) — 8-bit matrix operand order repaired and checked by reading, compiled, not run on hardware
//
#include <hip/hip_runtime.h>


typedef int   v8i __attribute__((ext_vector_type(8)));
typedef float v8f __attribute__((ext_vector_type(8)));
typedef float v2f __attribute__((ext_vector_type(2)));

#define N_TOK  8192
#define K_IN   4096
#define O_OUT  4096
#define R_RANK 32

#define BM 128
#define BN 128
#define BK 64
#define NKS (K_IN / BK)

__device__ __forceinline__ unsigned pack_nib2(int p0, int p1) {
  unsigned b0 = (unsigned)((( p0        & 0xF) ^ 8) - 8) & 0xFFu;
  unsigned b1 = (unsigned)((((p0 >> 4) & 0xF) ^ 8) - 8) & 0xFFu;
  unsigned b2 = (unsigned)((( p1        & 0xF) ^ 8) - 8) & 0xFFu;
  unsigned b3 = (unsigned)((((p1 >> 4) & 0xF) ^ 8) - 8) & 0xFFu;
  return b0 | (b1 << 8) | (b2 << 16) | (b3 << 24);
}

__device__ __forceinline__ unsigned pack_i32x4(int4 q) {
  return  ((unsigned)q.x & 0xFFu)
       | (((unsigned)q.y & 0xFFu) << 8)
       | (((unsigned)q.z & 0xFFu) << 16)
       | (((unsigned)q.w)         << 24);
}

__launch_bounds__(256, 1)
__global__ void w4a4_outlier_gemm(const int*   __restrict__ qx,
                                  const int*   __restrict__ wp,
                                  const float* __restrict__ sx,
                                  const float* __restrict__ wsc,
                                  const float* __restrict__ ox,
                                  const float* __restrict__ wo,
                                  const float* __restrict__ bias,
                                  float*       __restrict__ out) {
  __shared__ __align__(16) signed char sA[2][BM][BK];
  __shared__ __align__(16) signed char sB[2][BN][BK];
  __shared__ __align__(16) float       sOX[BM][R_RANK];
  __shared__ __align__(16) float       sWO[BN][R_RANK];

  const int tid  = threadIdx.x;
  const int lane = tid & 31;
  const int wave = tid >> 5;
  const int wm0  = (wave >> 2) * 64;
  const int wn0  = (wave & 3)  * 32;
  const int l15  = lane & 15;
  const int half = lane >> 4;

  const int m0 = blockIdx.y * BM;
  const int n0 = blockIdx.x * BN;

  {
    const float4* src_ox = (const float4*)(ox + (size_t)m0 * R_RANK);
    const float4* src_wo = (const float4*)(wo + (size_t)n0 * R_RANK);
    float4* dst_ox = (float4*)&sOX[0][0];
    float4* dst_wo = (float4*)&sWO[0][0];
#pragma unroll
    for (int j = 0; j < 4; ++j) {
      const int idx = tid + 256 * j;
      dst_ox[idx] = src_ox[idx];
      dst_wo[idx] = src_wo[idx];
    }
  }

  const int f_row = tid >> 1;
  const int f_kh  = (tid & 1) * 32;

  auto loadA = [&](int kk, unsigned va[8]) {
    const int4* p = (const int4*)(qx + (size_t)(m0 + f_row) * K_IN + kk + f_kh);
#pragma unroll
    for (int j = 0; j < 8; ++j) va[j] = pack_i32x4(p[j]);
  };
  auto loadB = [&](int kk, unsigned vb[8]) {
    const int4* p = (const int4*)(wp + (size_t)(n0 + f_row) * (K_IN / 2) + (kk + f_kh) / 2);
#pragma unroll
    for (int j = 0; j < 4; ++j) {
      const int4 q = p[j];
      vb[2 * j]     = pack_nib2(q.x, q.y);
      vb[2 * j + 1] = pack_nib2(q.z, q.w);
    }
  };
  auto storeA = [&](int buf, const unsigned va[8]) {
    uint4* d = (uint4*)&sA[buf][f_row][f_kh];
    d[0] = make_uint4(va[0], va[1], va[2], va[3]);
    d[1] = make_uint4(va[4], va[5], va[6], va[7]);
  };
  auto storeB = [&](int buf, const unsigned vb[8]) {
    uint4* d = (uint4*)&sB[buf][f_row][f_kh];
    d[0] = make_uint4(vb[0], vb[1], vb[2], vb[3]);
    d[1] = make_uint4(vb[4], vb[5], vb[6], vb[7]);
  };

  v8i acc[4][2];
#pragma unroll
  for (int tm = 0; tm < 4; ++tm)
#pragma unroll
    for (int tn = 0; tn < 2; ++tn) acc[tm][tn] = (v8i)0;

  unsigned va[8], vb[8];
  loadA(0, va); loadB(0, vb);
  storeA(0, va); storeB(0, vb);
  __syncthreads();

  for (int ks = 0; ks < NKS; ++ks) {
    const int buf = ks & 1;

    if (ks + 2 < NKS) {
      __builtin_prefetch(qx + (size_t)(m0 + f_row) * K_IN + (ks + 2) * BK + f_kh, 0, 0);
      __builtin_prefetch(wp + (size_t)(n0 + f_row) * (K_IN / 2) + ((ks + 2) * BK + f_kh) / 2, 0, 0);
    }
    if (ks + 1 < NKS) { loadA((ks + 1) * BK, va); loadB((ks + 1) * BK, vb); }

    v8i afr[4];
#pragma unroll
    for (int tm = 0; tm < 4; ++tm) {
      const signed char* base = &sA[buf][wm0 + tm * 16 + l15][0];
      const int2 p0 = *(const int2*)(base +  0 + 16 * half);
      const int2 p1 = *(const int2*)(base +  8 + 16 * half);
      const int2 p2 = *(const int2*)(base + 32 + 16 * half);
      const int2 p3 = *(const int2*)(base + 40 + 16 * half);
      v8i a;
      a[0] = p0.x; a[1] = p0.y; a[2] = p1.x; a[3] = p1.y;
      a[4] = p2.x; a[5] = p2.y; a[6] = p3.x; a[7] = p3.y;
      afr[tm] = a;
    }
    v8i bfr[2];
#pragma unroll
    for (int tn = 0; tn < 2; ++tn) {
      const signed char* base = &sB[buf][wn0 + tn * 16 + l15][0];
      const int4 q0 = *(const int4*)(base +  0 + 16 * half);
      const int4 q1 = *(const int4*)(base + 32 + 16 * half);
      v8i b;
      b[0] = q0.x; b[1] = q0.y; b[2] = q0.z; b[3] = q0.w;
      b[4] = q1.x; b[5] = q1.y; b[6] = q1.z; b[7] = q1.w;
      bfr[tn] = b;
    }

#pragma unroll
    for (int tm = 0; tm < 4; ++tm)
#pragma unroll
      for (int tn = 0; tn < 2; ++tn)
        acc[tm][tn] = __builtin_amdgcn_wmma_i32_16x16x64_iu8(
             true, afr[tm],  true, bfr[tn],
            acc[tm][tn],  false,  false);

    __syncthreads();
    if (ks + 1 < NKS) { storeA(buf ^ 1, va); storeB(buf ^ 1, vb); }
    __syncthreads();
  }

  float4 sxlo[4], sxhi[4];
#pragma unroll
  for (int tm = 0; tm < 4; ++tm) {
    const float* s = sx + m0 + wm0 + tm * 16 + 8 * half;
    sxlo[tm] = *(const float4*)(s);
    sxhi[tm] = *(const float4*)(s + 4);
  }
  float swv[2], bv[2];
#pragma unroll
  for (int tn = 0; tn < 2; ++tn) {
    swv[tn] = wsc[n0 + wn0 + tn * 16 + l15];
    bv[tn]  = bias[n0 + wn0 + tn * 16 + l15];
  }

#pragma unroll
  for (int tm = 0; tm < 4; ++tm) {
#pragma unroll
    for (int tn = 0; tn < 2; ++tn) {
      v8f facc = (v8f)0.0f;
#pragma unroll
      for (int k0 = 0; k0 < R_RANK; k0 += 4) {
        const float* ap = &sOX[wm0 + tm * 16 + l15][k0 + 2 * half];
        const float* bp = &sWO[wn0 + tn * 16 + l15][k0 + 2 * half];
        v2f a2; a2[0] = ap[0]; a2[1] = ap[1];
        v2f b2; b2[0] = bp[0]; b2[1] = bp[1];
        facc = __builtin_amdgcn_wmma_f32_16x16x4_f32(
            false, a2, false, b2, (short)0, facc, false, false);
      }
#pragma unroll
      for (int i = 0; i < 8; ++i) {
        const float sxi = (i < 4) ? ((const float*)&sxlo[tm])[i]
                                  : ((const float*)&sxhi[tm])[i - 4];
        const float v = (float)acc[tm][tn][i] * sxi * swv[tn] + facc[i] + bv[tn];
        const int grow = m0 + wm0 + tm * 16 + i + 8 * half;
        const int gcol = n0 + wn0 + tn * 16 + l15;
        out[(size_t)grow * O_OUT + gcol] = v;
      }
    }
  }
}

extern "C" void kernel_launch(void* const* d_in, const int* in_sizes, int n_in,
                              void* d_out, int out_size, void* d_ws, size_t ws_size,
                              hipStream_t stream) {
  const int*   quant_x       = (const int*)  d_in[0];
  const int*   weight_packed = (const int*)  d_in[1];
  const float* scale_x       = (const float*)d_in[2];
  const float* weight_scales = (const float*)d_in[3];
  const float* outlier_x     = (const float*)d_in[4];
  const float* w_outlier     = (const float*)d_in[5];
  const float* bias          = (const float*)d_in[6];
  float*       out           = (float*)d_out;

  dim3 grid(O_OUT / BN, N_TOK / BM);
  dim3 block(256);
  w4a4_outlier_gemm<<<grid, block, 0, stream>>>(
      quant_x, weight_packed, scale_x, weight_scales,
      outlier_x, w_outlier, bias, out);
}
